// RGCNLayer_82678120448330
// MI455X (gfx1250) — matrix operand order repaired and checked by reading, not compiled, not run on hardware
//
#include <hip/hip_runtime.h>
#include <stdint.h>

typedef __attribute__((ext_vector_type(16))) __bf16 v16bf;
typedef __attribute__((ext_vector_type(8)))  float  v8f;

#define TDIM    128
#define NREL    16
#define NBASES  8
#define KTOT    (17 * 128)   // 16 relations + self
#define NKB     (KTOT / 32)  // 68 K-blocks of 32
#define SELF_SLAB 16

// round-to-nearest-even f32 -> bf16 (bit trick; inputs are finite)
__device__ __forceinline__ unsigned short f2bf(float f) {
    unsigned x = __float_as_uint(f);
    unsigned r = ((x >> 16) & 1u) + 0x7fffu;
    return (unsigned short)((x + r) >> 16);
}

// ---------------------------------------------------------------------------
// Kernel 1: build W_all = [W_0..W_15 ; W_self] (2176 x 128) in bf16, packed in
// this file's contiguous-K WMMA B-fragment layout: block(kb,nt) is 1KB; lane L owns 32 bytes
// = Wfull[kb*32 + (L&16) + e][nt*16 + (L&15)], e = 0..15.
// ---------------------------------------------------------------------------
__global__ void pack_w_kernel(const float* __restrict__ bases,
                              const float* __restrict__ coeff,
                              const float* __restrict__ wself,
                              unsigned short* __restrict__ wpack) {
    int idx = blockIdx.x * 256 + threadIdx.x;      // 68*4096 = 278528 elems
    int e   = idx & 15;
    int L   = (idx >> 4) & 31;
    int nt  = (idx >> 9) & 7;
    int kb  = idx >> 12;
    if (kb >= NKB) return;
    int n  = nt * 16 + (L & 15);
    int kd = (L & 16) + e;                         // lanes 0-15: K 0..15, 16-31: K 16..31
    int k  = kb * 32 + kd;
    float v;
    if (k < NREL * TDIM) {
        int r = k >> 7, i = k & 127;
        float acc = 0.f;
#pragma unroll
        for (int b = 0; b < NBASES; b++)
            acc += coeff[r * NBASES + b] * bases[((size_t)b * TDIM + i) * TDIM + n];
        v = acc;
    } else {
        int i = k - NREL * TDIM;
        v = wself[i * TDIM + n];
    }
    wpack[idx] = f2bf(v);
}

// ---------------------------------------------------------------------------
// Kernel 2: x (f32) -> bf16 into slab 16 of A_all, zero-padding tail nodes.
// ---------------------------------------------------------------------------
__global__ void convert_x_kernel(const float* __restrict__ x,
                                 unsigned short* __restrict__ xslab,
                                 int N, int Np) {
    int idx = blockIdx.x * 256 + threadIdx.x;
    if (idx >= Np * TDIM) return;
    int n = idx >> 7;
    float v = (n < N) ? x[idx] : 0.f;
    xslab[idx] = f2bf(v);
}

// ---------------------------------------------------------------------------
// Kernel 3: per-(relation, target) degree counts.
// ---------------------------------------------------------------------------
__global__ void deg_kernel(const int* __restrict__ tgt, const int* __restrict__ rel,
                           int* __restrict__ deg, int E, int Np) {
    int e = blockIdx.x * 256 + threadIdx.x;
    if (e >= E) return;
    atomicAdd(&deg[(size_t)rel[e] * Np + tgt[e]], 1);
}

// ---------------------------------------------------------------------------
// Kernel 4: edge scatter. One wave per edge; each lane handles 4 dims.
// Adds x[src]/deg into agg[rel, tgt] via packed-bf16 global atomics, so agg
// ends up holding the per-(rel,node) MEAN directly.
// ---------------------------------------------------------------------------
__global__ void scatter_kernel(const float* __restrict__ x,
                               const int* __restrict__ src,
                               const int* __restrict__ tgt,
                               const int* __restrict__ rel,
                               const int* __restrict__ deg,
                               unsigned short* __restrict__ agg,
                               int E, int Np) {
    int eid  = blockIdx.x * 8 + (threadIdx.x >> 5);
    int lane = threadIdx.x & 31;
    if (eid >= E) return;
    int    s    = src[eid];
    size_t row  = (size_t)rel[eid] * Np + tgt[eid];
    int    d    = deg[row];
    float  scale = 1.0f / (float)(d > 0 ? d : 1);
    const float4 xv = *(const float4*)(x + (size_t)s * TDIM + lane * 4);
    unsigned p0 = (unsigned)f2bf(xv.x * scale) | ((unsigned)f2bf(xv.y * scale) << 16);
    unsigned p1 = (unsigned)f2bf(xv.z * scale) | ((unsigned)f2bf(xv.w * scale) << 16);
    unsigned long long addr = (unsigned long long)(agg + row * TDIM + lane * 4);
    asm volatile("global_atomic_pk_add_bf16 %0, %1, off" :: "v"(addr),        "v"(p0) : "memory");
    asm volatile("global_atomic_pk_add_bf16 %0, %1, off" :: "v"(addr + 4ull), "v"(p1) : "memory");
}

// ---------------------------------------------------------------------------
// Kernel 5: dense phase. hidden = relu( A_all_tile[16x2176] @ W_all[2176x128] )
// Wave32 WMMA bf16: each wave owns 32 nodes (2 M-tiles x 8 N-tiles of f32
// accumulators). W fragments are double-buffered through LDS, shared by the
// 8 waves of the block (256 nodes / block).
// ---------------------------------------------------------------------------
__global__ void __launch_bounds__(256)
rgcn_gemm_kernel(const unsigned short* __restrict__ A_all,
                 const unsigned short* __restrict__ wpack,
                 float* __restrict__ out, int N, int Np) {
    __shared__ uint4 ldsW[2][512];                 // 2 x 8KB (one K-block of W)
    const int t    = threadIdx.x;
    const int wave = t >> 5;
    const int lane = t & 31;
    const int m    = lane & 15;
    const int koff = (lane & 16) ? 16 : 0;         // contiguous K: lanes 0-15 K 0..15, 16-31 K 16..31
    const size_t nodeBase = (size_t)blockIdx.x * 256 + (size_t)wave * 32;

    union Frag { uint4 u[2]; v16bf v; };

    v8f c[2][8];
#pragma unroll
    for (int i = 0; i < 2; i++)
#pragma unroll
        for (int j = 0; j < 8; j++)
#pragma unroll
            for (int k = 0; k < 8; k++) c[i][j][k] = 0.f;

    const uint4* wsrc = (const uint4*)wpack;
    // stage K-block 0
    ldsW[0][t * 2]     = wsrc[t * 2];
    ldsW[0][t * 2 + 1] = wsrc[t * 2 + 1];
    __syncthreads();

    int buf = 0;
    for (int kb = 0; kb < NKB; kb++) {
        if (kb + 1 < NKB) {                        // prefetch next W block
            const uint4* s = wsrc + (size_t)(kb + 1) * 512;
            ldsW[buf ^ 1][t * 2]     = s[t * 2];
            ldsW[buf ^ 1][t * 2 + 1] = s[t * 2 + 1];
        }
        // A fragments: slab = kb/4 (relation 0..15, 16 = self), col = (kb%4)*32
        const unsigned short* aptr =
            A_all + (size_t)(kb >> 2) * Np * TDIM + (size_t)((kb & 3) * 32);
        Frag a0, a1;
        {
            const unsigned short* p0 = aptr + (nodeBase + m) * TDIM + koff;
            a0.u[0] = *(const uint4*)(p0);
            a0.u[1] = *(const uint4*)(p0 + 8);
            const unsigned short* p1 = aptr + (nodeBase + 16 + m) * TDIM + koff;
            a1.u[0] = *(const uint4*)(p1);
            a1.u[1] = *(const uint4*)(p1 + 8);
        }
#pragma unroll
        for (int nt = 0; nt < 8; nt++) {
            Frag b;
            const uint4* lp = &ldsW[buf][nt * 64 + lane * 2];
            b.u[0] = lp[0];
            b.u[1] = lp[1];
            c[0][nt] = __builtin_amdgcn_wmma_f32_16x16x32_bf16(
                false, a0.v, false, b.v, (short)0, c[0][nt], false, false);
            c[1][nt] = __builtin_amdgcn_wmma_f32_16x16x32_bf16(
                false, a1.v, false, b.v, (short)0, c[1][nt], false, false);
        }
        __syncthreads();
        buf ^= 1;
    }

    // store with ReLU; C/D layout: VGPR i -> row M = i + (lane>=16 ? 8 : 0), col N = lane&15
    const int nc = lane & 15;
    const int mo = (lane & 16) ? 8 : 0;
#pragma unroll
    for (int mt = 0; mt < 2; mt++) {
#pragma unroll
        for (int i = 0; i < 8; i++) {
            size_t node = nodeBase + (size_t)mt * 16 + mo + i;
            if (node < (size_t)N) {
                float* orow = out + node * TDIM;
#pragma unroll
                for (int nt = 0; nt < 8; nt++) {
                    float v = c[mt][nt][i];
                    orow[nt * 16 + nc] = v > 0.f ? v : 0.f;
                }
            }
        }
    }
}

// ---------------------------------------------------------------------------
extern "C" void kernel_launch(void* const* d_in, const int* in_sizes, int n_in,
                              void* d_out, int out_size, void* d_ws, size_t ws_size,
                              hipStream_t stream) {
    const float* x     = (const float*)d_in[0];
    const int*   edge  = (const int*)d_in[1];
    const int*   rel   = (const int*)d_in[2];
    const float* bases = (const float*)d_in[3];
    const float* coeff = (const float*)d_in[4];
    const float* wself = (const float*)d_in[5];
    float*       out   = (float*)d_out;

    const int N  = in_sizes[0] / TDIM;   // 50000
    const int E  = in_sizes[2];          // 800000
    const int Np = ((N + 255) / 256) * 256;  // padded: 256 nodes per block

    const int* srcIdx = edge;
    const int* tgtIdx = edge + E;

    char* ws = (char*)d_ws;
    unsigned short* A_all = (unsigned short*)ws;             // 17 * Np * 128 bf16
    size_t aBytes = (size_t)17 * Np * TDIM * 2;
    int* deg = (int*)(ws + aBytes);                          // 16 * Np int32
    size_t dBytes = (size_t)NREL * Np * 4;
    unsigned short* wpack = (unsigned short*)(ws + aBytes + dBytes); // 544 KB

    // zero the 16 aggregation slabs (bf16 zero = 0x0000) and the degree counts
    hipMemsetAsync(A_all, 0, (size_t)NREL * Np * TDIM * 2, stream);
    hipMemsetAsync(deg, 0, dBytes, stream);

    pack_w_kernel<<<(NKB * 4096 + 255) / 256, 256, 0, stream>>>(bases, coeff, wself, wpack);
    convert_x_kernel<<<(unsigned)((Np * TDIM + 255) / 256), 256, 0, stream>>>(
        x, A_all + (size_t)SELF_SLAB * Np * TDIM, N, Np);
    deg_kernel<<<(E + 255) / 256, 256, 0, stream>>>(tgtIdx, rel, deg, E, Np);
    scatter_kernel<<<(E + 7) / 8, 256, 0, stream>>>(x, srcIdx, tgtIdx, rel, deg, A_all, E, Np);
    rgcn_gemm_kernel<<<Np / 256, 256, 0, stream>>>(A_all, wpack, out, N, Np);
}
